// DGC_Layer_35579509080586
// MI455X (gfx1250) — matrix operand order repaired and checked by reading, not compiled, not run on hardware
//
#include <hip/hip_runtime.h>

// ---------------------------------------------------------------------------
// DGC layer for MI455X (gfx1250, wave32):
//   HW  = H @ W
//   DAD = d^-1/2 * relu((A+A^T)/2) * d^-1/2       (stored as hi/lo bf16 pair)
//   out = relu(HW - DAD @ HW)
// Big GEMM: split-bf16 (hi/lo) emulated-fp32 via v_wmma_f32_16x16x32_bf16,
// operands staged with GLOBAL_LOAD_ASYNC_TO_LDS_B128 (ASYNCcnt), double-buffered.
// ---------------------------------------------------------------------------

typedef __attribute__((ext_vector_type(16))) __bf16 v16bf;
typedef __attribute__((ext_vector_type(8)))  float  v8f;
typedef __attribute__((ext_vector_type(4)))  float  f32x4;

#define BM 128
#define BN 128
#define BK 32
#define LDS_STRIDE 40  // BK + 8 pad ushorts -> 80B rows (16B aligned)

union FragBF {
  f32x4 q[2];
  v16bf v;
};

__device__ __forceinline__ void split_bf16(float x, unsigned short& hi, unsigned short& lo) {
  // RNE float -> bf16 hi, residual -> bf16 lo  (a ~= hi + lo, ~16 mantissa bits)
  unsigned int u  = __float_as_uint(x);
  unsigned int rh = u + 0x7FFFu + ((u >> 16) & 1u);
  unsigned short h = (unsigned short)(rh >> 16);
  float hf = __uint_as_float((unsigned int)h << 16);
  float l  = x - hf;
  unsigned int ul = __float_as_uint(l);
  unsigned int rl = ul + 0x7FFFu + ((ul >> 16) & 1u);
  hi = h;
  lo = (unsigned short)(rl >> 16);
}

__device__ __forceinline__ v8f wmma_bf16(v16bf a, v16bf b, v8f c) {
  return __builtin_amdgcn_wmma_f32_16x16x32_bf16(false, a, false, b, (short)0, c, false, false);
}

// CDNA5 async global->LDS copy, 16B per lane, tracked by ASYNCcnt.
__device__ __forceinline__ void async_b128(unsigned int lds_addr, const void* gptr) {
  asm volatile("global_load_async_to_lds_b128 %0, %1, off"
               :: "v"(lds_addr), "v"((unsigned long long)(size_t)gptr)
               : "memory");
}
__device__ __forceinline__ void wait_async0() {
  asm volatile("s_wait_asynccnt 0" ::: "memory");
}

// ---------------------------------------------------------------------------
// GEMM1 (small, f32 operands): C = A[M x K] * B[K x N], register-split staging.
// ---------------------------------------------------------------------------
__global__ __launch_bounds__(256) void gemm_bf16x3(
    const float* __restrict__ Ap, int lda,
    const float* __restrict__ Bp, int ldb,
    float* __restrict__ Cout, int ldc, int K) {
  __shared__ __align__(16) unsigned short sAhi[BM][LDS_STRIDE];
  __shared__ __align__(16) unsigned short sAlo[BM][LDS_STRIDE];
  __shared__ __align__(16) unsigned short sBhi[BN][LDS_STRIDE];  // transposed [n][k]
  __shared__ __align__(16) unsigned short sBlo[BN][LDS_STRIDE];

  const int tid  = threadIdx.x;
  const int lane = tid & 31;
  const int wave = tid >> 5;
  const int wm   = (wave & 3) * 32;
  const int wn   = (wave >> 2) * 64;
  const long long blockM = (long long)blockIdx.x * BM;
  const long long blockN = (long long)blockIdx.y * BN;

  const int fr    = lane & 15;
  const int khalf = lane >> 4;
  const int a_k0  = khalf * 8;
  const int b_k0  = khalf * 8;

  const v8f vzero = {0.f, 0.f, 0.f, 0.f, 0.f, 0.f, 0.f, 0.f};
  v8f acc[2][4];
#pragma unroll
  for (int i = 0; i < 2; ++i)
#pragma unroll
    for (int j = 0; j < 4; ++j) acc[i][j] = vzero;

  const int ar = tid >> 3;
  const int ac = (tid & 7) * 4;
  const int bk = tid >> 5;
  const int bc = (tid & 31) * 4;

  for (int kt = 0; kt < K; kt += BK) {
#pragma unroll
    for (int i = 0; i < 4; ++i) {
      const int row = ar + i * 32;
      f32x4 v = *(const f32x4*)(Ap + (blockM + row) * (long long)lda + kt + ac);
      unsigned long long ph = 0, pl = 0;
#pragma unroll
      for (int j = 0; j < 4; ++j) {
        unsigned short h, l;
        split_bf16(v[j], h, l);
        ph |= (unsigned long long)h << (16 * j);
        pl |= (unsigned long long)l << (16 * j);
      }
      *(unsigned long long*)&sAhi[row][ac] = ph;
      *(unsigned long long*)&sAlo[row][ac] = pl;
    }
#pragma unroll
    for (int i = 0; i < 4; ++i) {
      const int krow = bk + i * 8;
      f32x4 v = *(const f32x4*)(Bp + (long long)(kt + krow) * ldb + blockN + bc);
#pragma unroll
      for (int j = 0; j < 4; ++j) {
        unsigned short h, l;
        split_bf16(v[j], h, l);
        sBhi[bc + j][krow] = h;
        sBlo[bc + j][krow] = l;
      }
    }
    __syncthreads();

    FragBF ahi[2], alo[2], bhi[4], blo[4];
#pragma unroll
    for (int im = 0; im < 2; ++im) {
      const int r = wm + im * 16 + fr;
      ahi[im].q[0] = *(const f32x4*)&sAhi[r][a_k0];
      ahi[im].q[1] = *(const f32x4*)&sAhi[r][a_k0 + 16];
      alo[im].q[0] = *(const f32x4*)&sAlo[r][a_k0];
      alo[im].q[1] = *(const f32x4*)&sAlo[r][a_k0 + 16];
    }
#pragma unroll
    for (int in = 0; in < 4; ++in) {
      const int c = wn + in * 16 + fr;
      bhi[in].q[0] = *(const f32x4*)&sBhi[c][b_k0];
      bhi[in].q[1] = *(const f32x4*)&sBhi[c][b_k0 + 16];
      blo[in].q[0] = *(const f32x4*)&sBlo[c][b_k0];
      blo[in].q[1] = *(const f32x4*)&sBlo[c][b_k0 + 16];
    }
#pragma unroll
    for (int im = 0; im < 2; ++im)
#pragma unroll
      for (int in = 0; in < 4; ++in) {
        acc[im][in] = wmma_bf16(ahi[im].v, bhi[in].v, acc[im][in]);
        acc[im][in] = wmma_bf16(ahi[im].v, blo[in].v, acc[im][in]);
        acc[im][in] = wmma_bf16(alo[im].v, bhi[in].v, acc[im][in]);
      }
    __syncthreads();
  }

#pragma unroll
  for (int im = 0; im < 2; ++im) {
#pragma unroll
    for (int in = 0; in < 4; ++in) {
      const long long row0 = blockM + wm + im * 16 + khalf * 8;
      const long long col  = blockN + wn + in * 16 + fr;
#pragma unroll
      for (int j = 0; j < 8; ++j) Cout[(row0 + j) * ldc + col] = acc[im][in][j];
    }
  }
}

// ---------------------------------------------------------------------------
// GEMM2 (dominant): out = relu(Cin - A*B) with bf16 hi/lo operands in global,
// async-staged into double-buffered LDS. A: [MxK] row-major bf16 (hi/lo),
// B: transposed [NxK] row-major bf16 (hi/lo), batch via blockIdx.z.
// ---------------------------------------------------------------------------
__global__ __launch_bounds__(256) void gemm2_async(
    const unsigned short* __restrict__ Ahi, const unsigned short* __restrict__ Alo, int lda,
    const unsigned short* __restrict__ BhiP, const unsigned short* __restrict__ BloP, int ldb,
    long long bStride,
    const float* __restrict__ CinP, float* __restrict__ CoutP, long long cStride, int ldc,
    int K) {
  __shared__ __align__(16) unsigned short sAh[2][BM][LDS_STRIDE];
  __shared__ __align__(16) unsigned short sAl[2][BM][LDS_STRIDE];
  __shared__ __align__(16) unsigned short sBh[2][BN][LDS_STRIDE];
  __shared__ __align__(16) unsigned short sBl[2][BN][LDS_STRIDE];

  const unsigned short* Bhi = BhiP + (long long)blockIdx.z * bStride;
  const unsigned short* Blo = BloP + (long long)blockIdx.z * bStride;
  const float* Cin  = CinP  + (long long)blockIdx.z * cStride;
  float*       Cout = CoutP + (long long)blockIdx.z * cStride;

  const int tid  = threadIdx.x;
  const int lane = tid & 31;
  const int wave = tid >> 5;
  const int wm   = (wave & 3) * 32;
  const int wn   = (wave >> 2) * 64;
  const long long blockM = (long long)blockIdx.x * BM;
  const long long blockN = (long long)blockIdx.y * BN;

  const int fr    = lane & 15;
  const int khalf = lane >> 4;
  const int a_k0  = khalf * 8;
  const int b_k0  = khalf * 8;

  // async staging geometry: one b128 issue = 32 lanes x 16B = 8 rows x 64B
  const int srow   = (wave << 4) + (lane >> 2);  // + i*8, i in {0,1} -> 16 rows/wave
  const int schunk = (lane & 3) * 8;             // ushort offset within 32-elem row

  const v8f vzero = {0.f, 0.f, 0.f, 0.f, 0.f, 0.f, 0.f, 0.f};
  v8f acc[2][4];
#pragma unroll
  for (int i = 0; i < 2; ++i)
#pragma unroll
    for (int j = 0; j < 4; ++j) acc[i][j] = vzero;

  auto stage = [&](int buf, int kt) {
#pragma unroll
    for (int i = 0; i < 2; ++i) {
      const int r = srow + i * 8;
      async_b128((unsigned int)(size_t)&sAh[buf][r][schunk],
                 Ahi + (blockM + r) * (long long)lda + kt + schunk);
      async_b128((unsigned int)(size_t)&sAl[buf][r][schunk],
                 Alo + (blockM + r) * (long long)lda + kt + schunk);
      async_b128((unsigned int)(size_t)&sBh[buf][r][schunk],
                 Bhi + (blockN + r) * (long long)ldb + kt + schunk);
      async_b128((unsigned int)(size_t)&sBl[buf][r][schunk],
                 Blo + (blockN + r) * (long long)ldb + kt + schunk);
    }
  };

  const int KT = K / BK;
  stage(0, 0);
  for (int t = 0; t < KT; ++t) {
    wait_async0();      // this wave's staged tile is in LDS
    __syncthreads();    // everyone's tile is in LDS
    if (t + 1 < KT) stage((t + 1) & 1, (t + 1) * BK);
    const int buf = t & 1;

    FragBF ahi[2], alo[2], bhi[4], blo[4];
#pragma unroll
    for (int im = 0; im < 2; ++im) {
      const int r = wm + im * 16 + fr;
      ahi[im].q[0] = *(const f32x4*)&sAh[buf][r][a_k0];
      ahi[im].q[1] = *(const f32x4*)&sAh[buf][r][a_k0 + 16];
      alo[im].q[0] = *(const f32x4*)&sAl[buf][r][a_k0];
      alo[im].q[1] = *(const f32x4*)&sAl[buf][r][a_k0 + 16];
    }
#pragma unroll
    for (int in = 0; in < 4; ++in) {
      const int c = wn + in * 16 + fr;
      bhi[in].q[0] = *(const f32x4*)&sBh[buf][c][b_k0];
      bhi[in].q[1] = *(const f32x4*)&sBh[buf][c][b_k0 + 16];
      blo[in].q[0] = *(const f32x4*)&sBl[buf][c][b_k0];
      blo[in].q[1] = *(const f32x4*)&sBl[buf][c][b_k0 + 16];
    }
#pragma unroll
    for (int im = 0; im < 2; ++im)
#pragma unroll
      for (int in = 0; in < 4; ++in) {
        acc[im][in] = wmma_bf16(ahi[im].v, bhi[in].v, acc[im][in]);
        acc[im][in] = wmma_bf16(ahi[im].v, blo[in].v, acc[im][in]);
        acc[im][in] = wmma_bf16(alo[im].v, bhi[in].v, acc[im][in]);
      }
    __syncthreads();    // all reads of buf done before it is restaged
  }

#pragma unroll
  for (int im = 0; im < 2; ++im) {
#pragma unroll
    for (int in = 0; in < 4; ++in) {
      const long long row0 = blockM + wm + im * 16 + khalf * 8;
      const long long col  = blockN + wn + in * 16 + fr;
#pragma unroll
      for (int j = 0; j < 8; ++j) {
        const long long idx = (row0 + j) * ldc + col;
        float r = Cin[idx] - acc[im][in][j];
        Cout[idx] = r > 0.f ? r : 0.f;
      }
    }
  }
}

// ---------------------------------------------------------------------------
// Prep kernels
// ---------------------------------------------------------------------------
__global__ void zero_f32(float* __restrict__ p, int n) {
  const int i = blockIdx.x * blockDim.x + threadIdx.x;
  if (i < n) p[i] = 0.f;
}

__global__ __launch_bounds__(256) void sym_rowsum(const float* __restrict__ A,
                                                  float* __restrict__ D, int N) {
  __shared__ float T[32][33];
  const int bj = blockIdx.x, bi = blockIdx.y;
  const int tx = threadIdx.x & 31, ty = threadIdx.x >> 5;
  for (int r = ty; r < 32; r += 8)
    T[r][tx] = A[(long long)(bj * 32 + r) * N + bi * 32 + tx];
  __syncthreads();
  for (int r = ty; r < 32; r += 8) {
    float a  = A[(long long)(bi * 32 + r) * N + bj * 32 + tx];
    float s  = 0.5f * (a + T[tx][r]);
    s = s > 0.f ? s : 0.f;
    for (int off = 16; off > 0; off >>= 1) s += __shfl_down(s, off, 32);
    if (tx == 0) atomicAdd(&D[bi * 32 + r], s);
  }
}

__global__ void inv_sqrt_deg(const float* __restrict__ D, float* __restrict__ dis, int n) {
  const int i = blockIdx.x * blockDim.x + threadIdx.x;
  if (i < n) {
    const float d = D[i];
    dis[i] = d > 0.f ? rsqrtf(d) : 0.f;
  }
}

// DAD[n][m] = dis[n]*relu((A[n][m]+A[m][n])/2)*dis[m], stored as hi/lo bf16
__global__ __launch_bounds__(256) void build_dad(const float* __restrict__ A,
                                                 const float* __restrict__ dis,
                                                 unsigned short* __restrict__ Dhi,
                                                 unsigned short* __restrict__ Dlo, int N) {
  __shared__ float T[32][33];
  const int bj = blockIdx.x, bi = blockIdx.y;
  const int tx = threadIdx.x & 31, ty = threadIdx.x >> 5;
  for (int r = ty; r < 32; r += 8)
    T[r][tx] = A[(long long)(bj * 32 + r) * N + bi * 32 + tx];
  __syncthreads();
  const float dj = dis[bj * 32 + tx];
  for (int r = ty; r < 32; r += 8) {
    float a = A[(long long)(bi * 32 + r) * N + bj * 32 + tx];
    float s = 0.5f * (a + T[tx][r]);
    s = s > 0.f ? s : 0.f;
    s = dis[bi * 32 + r] * s * dj;
    unsigned short h, l;
    split_bf16(s, h, l);
    const long long idx = (long long)(bi * 32 + r) * N + bj * 32 + tx;
    Dhi[idx] = h;
    Dlo[idx] = l;
  }
}

// HWT[b][o][n] (hi/lo bf16) from HW[b][n][o] f32 (tile transpose + split)
__global__ __launch_bounds__(256) void trans_split(const float* __restrict__ HW,
                                                   unsigned short* __restrict__ Thi,
                                                   unsigned short* __restrict__ Tlo) {
  __shared__ float T[32][33];
  const int b = blockIdx.z;
  const int n0 = blockIdx.x * 32, o0 = blockIdx.y * 32;
  const int tx = threadIdx.x & 31, ty = threadIdx.x >> 5;
  const float* src = HW + (size_t)b * 4096 * 256;
  for (int r = ty; r < 32; r += 8)
    T[r][tx] = src[(size_t)(n0 + r) * 256 + o0 + tx];
  __syncthreads();
  unsigned short* dhi = Thi + (size_t)b * 256 * 4096;
  unsigned short* dlo = Tlo + (size_t)b * 256 * 4096;
  for (int r = ty; r < 32; r += 8) {
    unsigned short h, l;
    split_bf16(T[tx][r], h, l);
    dhi[(size_t)(o0 + r) * 4096 + n0 + tx] = h;
    dlo[(size_t)(o0 + r) * 4096 + n0 + tx] = l;
  }
}

// ---------------------------------------------------------------------------
extern "C" void kernel_launch(void* const* d_in, const int* in_sizes, int n_in,
                              void* d_out, int out_size, void* d_ws, size_t ws_size,
                              hipStream_t stream) {
  (void)in_sizes; (void)n_in; (void)out_size; (void)ws_size;
  const float* H  = (const float*)d_in[0];  // [8,4096,256]
  const float* Wm = (const float*)d_in[1];  // [256,256]
  const float* A  = (const float*)d_in[2];  // [4096,4096]
  float* out = (float*)d_out;               // [8,4096,256]

  const int N = 4096, Bb = 8, F = 256;
  const size_t NN = (size_t)N * N, BNF = (size_t)Bb * N * F;

  char* ws = (char*)d_ws;
  float*          HW    = (float*)ws;                            // 32 MiB f32
  unsigned short* DADhi = (unsigned short*)(ws + BNF * 4);       // 32 MiB bf16
  unsigned short* DADlo = DADhi + NN;                            // 32 MiB bf16
  unsigned short* HWThi = DADlo + NN;                            // 16 MiB bf16
  unsigned short* HWTlo = HWThi + BNF;                           // 16 MiB bf16
  float*          Dd    = (float*)(HWTlo + BNF);
  float*          dis   = Dd + N;

  // adjacency normalization -> DAD (hi/lo bf16)
  zero_f32<<<(N + 255) / 256, 256, 0, stream>>>(Dd, N);
  sym_rowsum<<<dim3(N / 32, N / 32), 256, 0, stream>>>(A, Dd, N);
  inv_sqrt_deg<<<(N + 255) / 256, 256, 0, stream>>>(Dd, dis, N);
  build_dad<<<dim3(N / 32, N / 32), 256, 0, stream>>>(A, dis, DADhi, DADlo, N);

  // HW = H @ W : M = 32768, N = 256, K = 256
  gemm_bf16x3<<<dim3((Bb * N) / BM, F / BN, 1), 256, 0, stream>>>(H, F, Wm, F, HW, F, F);

  // transposed hi/lo split of HW for async B staging
  trans_split<<<dim3(N / 32, F / 32, Bb), 256, 0, stream>>>(HW, HWThi, HWTlo);

  // out = relu(HW - DAD @ HW), per batch: M = 4096, N = 256, K = 4096
  gemm2_async<<<dim3(N / BM, F / BN, Bb), 256, 0, stream>>>(
      DADhi, DADlo, N, HWThi, HWTlo, N, (long long)F * N,
      HW, out, (long long)N * F, F, N);
}
